// FISA_30167850287507
// MI455X (gfx1250) — matrix operand order repaired and checked by reading, not compiled, not run on hardware
//
#include <hip/hip_runtime.h>
#include <hip/hip_bf16.h>

// ---------------------------------------------------------------------------
// FISA stock model, CDNA5 (gfx1250) implementation.
//   Kernel 1: Ux = x @ Uall_w^T + Uall_b   (per-stock 4096x256x768 GEMM)
//             bf16 WMMA (v_wmma_f32_16x16x32_bf16), f32 accumulate,
//             double-buffered GLOBAL_LOAD_ASYNC_TO_LDS_B128 tile staging
//             (ASYNCcnt) when the toolchain exposes the builtin.
//   Kernel 2: per-stock TimeLSTM scan + day LSTM + output head, with all
//             matmuls (c@Wd^T, h@Wall^T, dv@lWih^T) done via WMMA on
//             LDS-resident bf16 weights and f32 states.
// Workspace: Ux buffer, 16*4096*256 f32 = 67,108,864 bytes in d_ws.
// ---------------------------------------------------------------------------

typedef __attribute__((ext_vector_type(16))) __bf16 v16bf;
typedef __attribute__((ext_vector_type(8)))  __bf16 v8bf;
typedef __attribute__((ext_vector_type(8)))  float  v8f;

#define NS 16
#define ND 64
#define NT 64
#define NE 768
#define NH 64
#define NG 256        // 4*H
#define NM 4096       // ND*NT

#define LDW 96        // bf16 LDS row stride (weights), 192B, 16B-aligned
#define LDF 68        // f32 LDS row stride (gemm tiles / states), 272B, 16B-aligned
#define LDH 68        // f32 LDS row stride (state matrices)
#define LDTS 65       // f32 LDS row stride for time inputs

#if __has_builtin(__builtin_amdgcn_global_load_async_to_lds_b128)
#define FISA_ASYNC_COPY 1
#else
#define FISA_ASYNC_COPY 0
#endif

#if FISA_ASYNC_COPY
// The builtin's params are pointers to 16-byte int vectors (GCC vector_size
// spelling, per the clang prototype), global (AS1) src and LDS (AS3) dst.
typedef int fisa_b128 __attribute__((vector_size(16)));
typedef __attribute__((address_space(1))) fisa_b128 fisa_b128_g;
typedef __attribute__((address_space(3))) fisa_b128 fisa_b128_l;

// Per-lane async DMA: 16 bytes global -> LDS, tracked by ASYNCcnt.
static __device__ __forceinline__ void async_cp16(const float* g, float* l) {
  __builtin_amdgcn_global_load_async_to_lds_b128(
      (fisa_b128_g*)g, (fisa_b128_l*)l, 0, 0);
}
static __device__ __forceinline__ void wait_async_zero() {
#if __has_builtin(__builtin_amdgcn_s_wait_asynccnt)
  __builtin_amdgcn_s_wait_asynccnt(0);
#else
  asm volatile("s_wait_asynccnt 0x0" ::: "memory");
#endif
}
#else
static __device__ __forceinline__ void wait_async_zero() {}
#endif

static __device__ __forceinline__ v8f vzero8() {
  v8f z;
#pragma unroll
  for (int i = 0; i < 8; ++i) z[i] = 0.0f;
  return z;
}

static __device__ __forceinline__ float sigf(float x) {
  return 1.0f / (1.0f + __expf(-x));
}

// A fragment (16x32, bf16). Lanes 0-15 hold row M=lane with
// K = kk..kk+15; lanes 16-31 hold row M=lane-16 with
// K = kk+16..kk+31.
static __device__ __forceinline__ v16bf ldsA_f32(const float* base, int ld, int kk, int lane) {
  const int half = lane >> 4;
  const int r = lane & 15;
  const float* p = base + r * ld + kk + half * 16;
  float4 a0 = *(const float4*)(p);
  float4 a1 = *(const float4*)(p + 4);
  float4 b0 = *(const float4*)(p + 8);
  float4 b1 = *(const float4*)(p + 12);
  v16bf o;
  o[0]=(__bf16)a0.x; o[1]=(__bf16)a0.y; o[2]=(__bf16)a0.z; o[3]=(__bf16)a0.w;
  o[4]=(__bf16)a1.x; o[5]=(__bf16)a1.y; o[6]=(__bf16)a1.z; o[7]=(__bf16)a1.w;
  o[8]=(__bf16)b0.x; o[9]=(__bf16)b0.y; o[10]=(__bf16)b0.z; o[11]=(__bf16)b0.w;
  o[12]=(__bf16)b1.x; o[13]=(__bf16)b1.y; o[14]=(__bf16)b1.z; o[15]=(__bf16)b1.w;
  return o;
}

// B fragment (32x16, bf16) from an f32 LDS tile. Lanes 0-15 hold column
// N=lane with K = kk..kk+15; lanes 16-31 hold column N=lane-16 with
// K = kk+16..kk+31. Weight rows (gate index) are contiguous in K.
static __device__ __forceinline__ v16bf ldsB_f32(const float* base, int ld, int kk, int lane) {
  const int half = lane >> 4;
  const int n = lane & 15;
  const float* p = base + n * ld + kk + half * 16;
  float4 a0 = *(const float4*)(p);
  float4 a1 = *(const float4*)(p + 4);
  float4 b0 = *(const float4*)(p + 8);
  float4 b1 = *(const float4*)(p + 12);
  v16bf o;
  o[0]=(__bf16)a0.x; o[1]=(__bf16)a0.y; o[2]=(__bf16)a0.z; o[3]=(__bf16)a0.w;
  o[4]=(__bf16)a1.x; o[5]=(__bf16)a1.y; o[6]=(__bf16)a1.z; o[7]=(__bf16)a1.w;
  o[8]=(__bf16)b0.x; o[9]=(__bf16)b0.y; o[10]=(__bf16)b0.z; o[11]=(__bf16)b0.w;
  o[12]=(__bf16)b1.x; o[13]=(__bf16)b1.y; o[14]=(__bf16)b1.z; o[15]=(__bf16)b1.w;
  return o;
}

// B fragment from a pre-converted bf16 LDS weight matrix (kernel 2).
static __device__ __forceinline__ v16bf ldsB_bf(const __bf16* base, int ld, int kk, int lane) {
  const int half = lane >> 4;
  const int n = lane & 15;
  const __bf16* p = base + n * ld + kk + half * 16;
  v8bf lo = *(const v8bf*)(p);
  v8bf hi = *(const v8bf*)(p + 8);
  v16bf o;
#pragma unroll
  for (int i = 0; i < 8; ++i) { o[i] = lo[i]; o[8 + i] = hi[i]; }
  return o;
}

// ---------------------------------------------------------------------------
// Kernel 1: Ux[s, m, g] = sum_e x[s, m, e] * Uall_w[s, g, e] + Uall_b[s, g]
// grid = (Mtiles=64, Ntiles=4, S=16), block = 256 (8 waves).
// Block tile: 64(M) x 64(N); K chunks of 64 double-buffered in LDS (f32),
// fetched with async LDS-direct DMA and converted to bf16 at fragment load
// (v_cvt_pk_bf16_f32 co-executes with the WMMA stream).
// ---------------------------------------------------------------------------
__global__ __launch_bounds__(256) void fisa_ux_gemm(
    const float* __restrict__ X, const float* __restrict__ Uw,
    const float* __restrict__ Ub, float* __restrict__ Ux)
{
  __shared__ alignas(16) float Abuf[2][64 * LDF];
  __shared__ alignas(16) float Bbuf[2][64 * LDF];

  const int s  = blockIdx.z;
  const int m0 = blockIdx.x * 64;
  const int n0 = blockIdx.y * 64;
  const int tid  = threadIdx.x;
  const int lane = tid & 31;
  const int w    = tid >> 5;       // wave 0..7
  const int tm   = w >> 1;         // M tile 0..3
  const int tn0  = (w & 1) * 2;    // N tiles {tn0, tn0+1}
  const int half = lane >> 4;
  const int col  = lane & 15;

  const float* Arow = X  + ((size_t)s * NM + m0) * NE;
  const float* Brow = Uw + ((size_t)s * NG + n0) * NE;

  // Fold the bias into the accumulator init (C/D layout: N = col, all VGPRs).
  v8f acc0, acc1;
  {
    float b0v = Ub[s * NG + n0 + (tn0 + 0) * 16 + col];
    float b1v = Ub[s * NG + n0 + (tn0 + 1) * 16 + col];
#pragma unroll
    for (int j = 0; j < 8; ++j) { acc0[j] = b0v; acc1[j] = b1v; }
  }

  const int lr   = tid >> 2;        // row 0..63
  const int lseg = (tid & 3) * 16;  // 16-float segment
  const float* pa = Arow + (size_t)lr * NE + lseg;
  const float* pb = Brow + (size_t)lr * NE + lseg;
  const int lofs = lr * LDF + lseg;

  // Stage one 64x64 f32 A tile + B tile into LDS buffer `buf`.
  auto stage = [&](int buf, int k0) {
#if FISA_ASYNC_COPY
#pragma unroll
    for (int i = 0; i < 16; i += 4) {
      async_cp16(pa + k0 + i, &Abuf[buf][lofs + i]);
      async_cp16(pb + k0 + i, &Bbuf[buf][lofs + i]);
    }
#else
#pragma unroll
    for (int i = 0; i < 16; i += 4) {
      *(float4*)&Abuf[buf][lofs + i] = *(const float4*)(pa + k0 + i);
      *(float4*)&Bbuf[buf][lofs + i] = *(const float4*)(pb + k0 + i);
    }
#endif
  };

  stage(0, 0);
  wait_async_zero();
  __syncthreads();

  int buf = 0;
  for (int k0 = 0; k0 < NE; k0 += 64, buf ^= 1) {
    const bool more = (k0 + 64 < NE);
    if (more) stage(buf ^ 1, k0 + 64);   // DMA next chunk while computing

#pragma unroll
    for (int kk = 0; kk < 64; kk += 32) {
      v16bf a  = ldsA_f32(&Abuf[buf][tm * 16 * LDF], LDF, kk, lane);
      v16bf b0 = ldsB_f32(&Bbuf[buf][(tn0 + 0) * 16 * LDF], LDF, kk, lane);
      v16bf b1 = ldsB_f32(&Bbuf[buf][(tn0 + 1) * 16 * LDF], LDF, kk, lane);
      acc0 = __builtin_amdgcn_wmma_f32_16x16x32_bf16(false, a, false, b0, (short)0, acc0, false, false);
      acc1 = __builtin_amdgcn_wmma_f32_16x16x32_bf16(false, a, false, b1, (short)0, acc1, false, false);
    }
    if (more) wait_async_zero();         // my DMA for buf^1 landed
    __syncthreads();                     // everyone done reading buf / writing buf^1
  }

  // C/D layout: VGPR j -> (M = 16*tm + 8*half + j, N = 16*tn + col)
  float* Obase = Ux + ((size_t)s * NM + m0) * NG + n0;
#pragma unroll
  for (int j = 0; j < 8; ++j) {
    int row = tm * 16 + half * 8 + j;
    Obase[(size_t)row * NG + (tn0 + 0) * 16 + col] = acc0[j];
    Obase[(size_t)row * NG + (tn0 + 1) * 16 + col] = acc1[j];
  }
}

// ---------------------------------------------------------------------------
// Kernel 2: one block per stock. TimeLSTM scan (t = 0..63, batch = 64 days),
// then day LSTM + output head. All matmuls via WMMA on LDS data.
// Wave w covers d-tile (w&3) and h-tiles {2*(w>>2), 2*(w>>2)+1} for ALL four
// gate blocks, so the LSTM gating combines entirely in registers.
// ---------------------------------------------------------------------------
__global__ __launch_bounds__(256) void fisa_recurrent(
    const float* __restrict__ Ux,   const float* __restrict__ TS,
    const float* __restrict__ Wd_w, const float* __restrict__ Wd_b,
    const float* __restrict__ Wall_w, const float* __restrict__ Wall_b,
    const float* __restrict__ lWih, const float* __restrict__ lbih,
    const float* __restrict__ lbhh,
    const float* __restrict__ ls_w, const float* __restrict__ ls_b,
    float* __restrict__ out)
{
  __shared__ alignas(16) float  h_s [64 * LDH];   // hidden state (D x H)
  __shared__ alignas(16) float  c_s [64 * LDH];   // cell state   (D x H)
  __shared__ alignas(16) float  dv_s[64 * LDH];   // day_vecs     (D x T)
  __shared__ alignas(16) float  ts_s[64 * LDTS];  // time inputs  (D x T)
  __shared__ alignas(16) __bf16 Wd_s[64 * LDW];   // Wd_w   (H x H)   bf16
  __shared__ alignas(16) __bf16 Wg_s[NG * LDW];   // Wall_w / lWih (4H x H) bf16
  __shared__ float bWd[NH];
  __shared__ float bWg[NG];
  __shared__ float lsw_s[ND];
  __shared__ float wred[8];

  const int s    = blockIdx.x;
  const int tid  = threadIdx.x;
  const int lane = tid & 31;
  const int w    = tid >> 5;
  const int half = lane >> 4;
  const int col  = lane & 15;
  const int tm   = w & 3;        // d tile 0..3
  const int hp   = w >> 2;       // h half: h tiles {2*hp, 2*hp+1}

  // ---- stage weights into LDS (f32 -> bf16), zero state ----
  for (int i = tid; i < NH * NH; i += 256) {
    int r = i >> 6, k = i & 63;
    Wd_s[r * LDW + k] = (__bf16)Wd_w[((size_t)s * NH + r) * NH + k];
    ts_s[r * LDTS + k] = TS[(size_t)s * NM + i];
  }
  for (int i = tid; i < NG * NH; i += 256) {
    int r = i >> 6, k = i & 63;
    Wg_s[r * LDW + k] = (__bf16)Wall_w[((size_t)s * NG + r) * NH + k];
  }
  for (int i = tid; i < NH; i += 256) { bWd[i] = Wd_b[s * NH + i]; lsw_s[i] = ls_w[i]; }
  for (int i = tid; i < NG; i += 256) bWg[i] = Wall_b[s * NG + i];
  for (int i = tid; i < 64 * LDH; i += 256) { h_s[i] = 0.f; c_s[i] = 0.f; dv_s[i] = 0.f; }
  __syncthreads();

  const float* UxS = Ux + (size_t)s * NM * NG;

  for (int t = 0; t < NT; ++t) {
    // ---- stage 1: c_s1 = tanh(c @ Wd^T + Wd_b); c_adj in registers ----
    v8f acc1[2];
    acc1[0] = vzero8(); acc1[1] = vzero8();
#pragma unroll
    for (int kk = 0; kk < NH; kk += 32) {
      v16bf a = ldsA_f32(c_s + tm * 16 * LDH, LDH, kk, lane);
#pragma unroll
      for (int it = 0; it < 2; ++it) {
        v16bf b = ldsB_bf(Wd_s + (hp * 2 + it) * 16 * LDW, LDW, kk, lane);
        acc1[it] = __builtin_amdgcn_wmma_f32_16x16x32_bf16(false, a, false, b, (short)0, acc1[it], false, false);
      }
    }
    float cadj[2][8];
#pragma unroll
    for (int it = 0; it < 2; ++it)
#pragma unroll
      for (int j = 0; j < 8; ++j) {
        int d  = tm * 16 + half * 8 + j;
        int hI = (hp * 2 + it) * 16 + col;
        float cs1 = tanhf(acc1[it][j] + bWd[hI]);
        cadj[it][j] = c_s[d * LDH + hI] - cs1 + cs1 * ts_s[d * LDTS + t];
      }

    // ---- stage 2: gates = sigmoid(h @ Wall^T + Wall_b + Ux[d,t,:]) ----
    v8f acc[4][2];
#pragma unroll
    for (int gb = 0; gb < 4; ++gb)
#pragma unroll
      for (int it = 0; it < 2; ++it) acc[gb][it] = vzero8();
#pragma unroll
    for (int kk = 0; kk < NH; kk += 32) {
      v16bf a = ldsA_f32(h_s + tm * 16 * LDH, LDH, kk, lane);
#pragma unroll
      for (int gb = 0; gb < 4; ++gb)
#pragma unroll
        for (int it = 0; it < 2; ++it) {
          int tn = gb * 4 + hp * 2 + it;
          v16bf b = ldsB_bf(Wg_s + tn * 16 * LDW, LDW, kk, lane);
          acc[gb][it] = __builtin_amdgcn_wmma_f32_16x16x32_bf16(false, a, false, b, (short)0, acc[gb][it], false, false);
        }
    }
#pragma unroll
    for (int gb = 0; gb < 4; ++gb)
#pragma unroll
      for (int it = 0; it < 2; ++it) {
        int tn = gb * 4 + hp * 2 + it;
#pragma unroll
        for (int j = 0; j < 8; ++j) {
          int d = tm * 16 + half * 8 + j;
          int g = tn * 16 + col;
          acc[gb][it][j] = sigf(acc[gb][it][j] + bWg[g] + UxS[((size_t)d * NT + t) * NG + g]);
        }
      }
    __syncthreads();   // all reads of h_s / c_s for this step are done

    // ---- elementwise update; source gate order: f, i, o, ct ----
#pragma unroll
    for (int it = 0; it < 2; ++it)
#pragma unroll
      for (int j = 0; j < 8; ++j) {
        int d  = tm * 16 + half * 8 + j;
        int hI = (hp * 2 + it) * 16 + col;
        float fg = acc[0][it][j], ig = acc[1][it][j];
        float og = acc[2][it][j], cg = acc[3][it][j];
        float cn = fg * cadj[it][j] + ig * cg;
        c_s[d * LDH + hI] = cn;
        h_s[d * LDH + hI] = og * tanhf(cn);
        // day_vecs[d][t] += sum over 16 h columns of the output gate
        float sum = og;
        sum += __shfl_xor(sum, 1);
        sum += __shfl_xor(sum, 2);
        sum += __shfl_xor(sum, 4);
        sum += __shfl_xor(sum, 8);
        if (col == 0) atomicAdd(&dv_s[d * LDH + t], sum);  // 4 adds/loc (2 waves x 2 h-tiles)
      }
    __syncthreads();   // updates visible to next step
  }

  // ---- day LSTM (seq_len=1, c0=0) + output head; PyTorch order i,f,g,o ----
  for (int i = tid; i < NG * NH; i += 256) {
    int r = i >> 6, k = i & 63;
    Wg_s[r * LDW + k] = (__bf16)lWih[((size_t)s * NG + r) * NH + k];
  }
  for (int i = tid; i < NG; i += 256) bWg[i] = lbih[s * NG + i] + lbhh[s * NG + i];
  __syncthreads();

  const int gbs[3] = {0, 2, 3};    // need only i, g, o (f multiplies c0 = 0)
  v8f acc2[3][2];
#pragma unroll
  for (int q = 0; q < 3; ++q)
#pragma unroll
    for (int it = 0; it < 2; ++it) acc2[q][it] = vzero8();
#pragma unroll
  for (int kk = 0; kk < NH; kk += 32) {
    v16bf a = ldsA_f32(dv_s + tm * 16 * LDH, LDH, kk, lane);
#pragma unroll
    for (int q = 0; q < 3; ++q)
#pragma unroll
      for (int it = 0; it < 2; ++it) {
        int tn = gbs[q] * 4 + hp * 2 + it;
        v16bf b = ldsB_bf(Wg_s + tn * 16 * LDW, LDW, kk, lane);
        acc2[q][it] = __builtin_amdgcn_wmma_f32_16x16x32_bf16(false, a, false, b, (short)0, acc2[q][it], false, false);
      }
  }

  float local = 0.f;
#pragma unroll
  for (int it = 0; it < 2; ++it)
#pragma unroll
    for (int j = 0; j < 8; ++j) {
      int d  = tm * 16 + half * 8 + j;
      int hI = (hp * 2 + it) * 16 + col;
      float ig = sigf (acc2[0][it][j] + bWg[hI]);          // i block: g = 0*64 + hI
      float gg = tanhf(acc2[1][it][j] + bWg[128 + hI]);    // g block
      float og = sigf (acc2[2][it][j] + bWg[192 + hI]);    // o block
      float hd = og * tanhf(ig * gg);
      local += hd * lsw_s[d];                              // ft[d]*ls_w[d] contribution
    }
  local += __shfl_xor(local, 1);
  local += __shfl_xor(local, 2);
  local += __shfl_xor(local, 4);
  local += __shfl_xor(local, 8);
  local += __shfl_xor(local, 16);
  if (lane == 0) wred[w] = local;
  __syncthreads();
  if (tid == 0) {
    float v = ls_b[0];
    for (int i = 0; i < 8; ++i) v += wred[i];    // fixed order: deterministic
    out[s] = v > 0.f ? v : 0.01f * v;
  }
}

// ---------------------------------------------------------------------------
extern "C" void kernel_launch(void* const* d_in, const int* in_sizes, int n_in,
                              void* d_out, int out_size, void* d_ws, size_t ws_size,
                              hipStream_t stream) {
  (void)in_sizes; (void)n_in; (void)out_size;
  const float* X      = (const float*)d_in[0];   // (S,D,T,E)
  const float* TS     = (const float*)d_in[1];   // (S,D,T)
  const float* Wd_w   = (const float*)d_in[2];   // (S,H,H)
  const float* Wd_b   = (const float*)d_in[3];   // (S,H)
  const float* Wall_w = (const float*)d_in[4];   // (S,4H,H)
  const float* Wall_b = (const float*)d_in[5];   // (S,4H)
  const float* Uall_w = (const float*)d_in[6];   // (S,4H,E)
  const float* Uall_b = (const float*)d_in[7];   // (S,4H)
  const float* lWih   = (const float*)d_in[8];   // (S,4H,H)
  const float* lbih   = (const float*)d_in[9];   // (S,4H)
  /* d_in[10] lWhh: only its bias matters (h0 = 0) */
  const float* lbhh   = (const float*)d_in[11];  // (S,4H)
  /* d_in[12..23]: attention params -> dead code (softmax over size-1 axis) */
  const float* ls_w   = (const float*)d_in[24];  // (1,H)
  const float* ls_b   = (const float*)d_in[25];  // (1,)
  float* out = (float*)d_out;                    // (S,1) f32

  // Workspace: Ux (S, D*T, 4H) f32 = 67,108,864 bytes.
  float* Ux = (float*)d_ws;
  (void)ws_size;

  dim3 g1(NM / 64, NG / 64, NS);   // (64, 4, 16)
  fisa_ux_gemm<<<g1, 256, 0, stream>>>(X, Uall_w, Uall_b, Ux);

  fisa_recurrent<<<dim3(NS), 256, 0, stream>>>(Ux, TS, Wd_w, Wd_b, Wall_w, Wall_b,
                                               lWih, lbih, lbhh, ls_w, ls_b, out);
}
